// MOE_83202106458680
// MI455X (gfx1250) — matrix operand order repaired and checked by reading, not compiled, not run on hardware
//
#include <hip/hip_runtime.h>
#include <hip/hip_bf16.h>

// ---------------- problem constants ----------------
#define C_DIM 768
#define H_DIM 3072
#define N_EXP 8
#define N_TOK 8192            // B*T = 4*2048
#define M_TOK 32              // tokens per block (2 wmma N-tiles of 16)

typedef unsigned short u16;
typedef __attribute__((ext_vector_type(16))) __bf16 v16bf;
typedef __attribute__((ext_vector_type(8)))  float  v8f;

// ---------------- LDS layout ----------------
#define XT_LD    40                              // halves per transposed row (32 + 8 pad, 80B: 16B aligned)
#define XS_OFF   0
#define XS_SZ    (C_DIM * XT_LD * 2)             // 61440  : x^T  [768][40]
#define HS_OFF   (XS_OFF + XS_SZ)
#define HS_SZ    (H_DIM * XT_LD * 2)             // 245760 : h^T  [3072][40]
#define TK_OFF   (HS_OFF + HS_SZ)
#define TK_SZ    (M_TOK * 4)
#define GV_OFF   (TK_OFF + TK_SZ)
#define GV_SZ    (M_TOK * 4)
#define SMEM_TOTAL (GV_OFF + GV_SZ)              // 307456 bytes (< 320KB WGP LDS)

// ---------------- workspace layout (bytes) ----------------
#define WS_XB      0                             // N_TOK*C bf16      = 12,582,912
#define WS_WFCB    12582912                      // E*H*C bf16        = 37,748,736
#define WS_WPJB    50331648                      // E*C*H bf16        = 37,748,736
#define WS_LISTS   88080384                      // E*N_TOK int       =    262,144
#define WS_GATES   88342528                      // E*N_TOK float     =    262,144
#define WS_COUNTS  88604672                      // E ints (padded)

// ---------------- helpers ----------------
__device__ inline u16 f2bf(float f) {            // fp32 -> bf16 RNE
    unsigned u = __builtin_bit_cast(unsigned, f);
    unsigned r = u + 0x7FFFu + ((u >> 16) & 1u);
    return (u16)(r >> 16);
}

// Branch-free tanh-approx gelu: tanh(u) = 1 - 2/(exp(2u)+1) via hardware v_exp_f32.
// Correct at both tails (exp->inf => 1, exp->0 => -1); no libdevice branches.
__device__ inline float gelu_fn(float x) {
    float u = 0.7978845608028654f * (x + 0.044715f * x * x * x);
    float t = 1.0f - 2.0f / (__expf(2.0f * u) + 1.0f);
    return 0.5f * x * (1.0f + t);
}

__device__ inline v8f wmma_bf16(v16bf a, v16bf b, v8f c) {
    return __builtin_amdgcn_wmma_f32_16x16x32_bf16(false, a, false, b, (short)0, c,
                                                   false, false);
}

// A fragment (16x32 bf16, M=weight rows, K contiguous in memory) loaded straight
// from global: lanes 0-15 row=lane K{0-7,16-23}; lanes 16-31 row=lane-16 K{8-15,24-31}
__device__ inline v16bf load_a_w(const u16* __restrict__ w, int ld, int row0, int k0,
                                 int lane) {
    const u16* p = w + (size_t)(row0 + (lane & 15)) * ld + k0 + ((lane < 16) ? 0 : 8);
    union { uint4 u[2]; v16bf v; } U;
    U.u[0] = *(const uint4*)p;         // K +0..7
    U.u[1] = *(const uint4*)(p + 16);  // K +16..23
    return U.v;
}

// B fragment (32x16 bf16, K x N=tokens) from LDS-transposed activations:
// lanes 0-15 token=m0+lane K{0-7,16-23}; lanes 16-31 token=m0+lane-16 K{8-15,24-31}.
__device__ inline v16bf load_b_t(const u16* bT, int k0, int m0, int lane) {
    const u16* p = bT + (size_t)(k0 + ((lane < 16) ? 0 : 8)) * XT_LD + m0 + (lane & 15);
    union { u16 s[16]; v16bf v; } U;
#pragma unroll
    for (int i = 0; i < 8; ++i) {
        U.s[i]     = p[i * XT_LD];
        U.s[8 + i] = p[(16 + i) * XT_LD];
    }
    return U.v;
}

// ---------------- kernels ----------------
__global__ void moe_zero_kernel(float* __restrict__ out, int n, int* __restrict__ counts) {
    if (blockIdx.x == 0 && threadIdx.x < N_EXP) counts[threadIdx.x] = 0;
    float4 z = make_float4(0.f, 0.f, 0.f, 0.f);
    for (int i = (blockIdx.x * 256 + threadIdx.x) * 4; i < n; i += gridDim.x * 256 * 4)
        *(float4*)(out + i) = z;
}

__global__ void cast_f32_bf16(const float* __restrict__ src, u16* __restrict__ dst, int n) {
    for (int i = (blockIdx.x * blockDim.x + threadIdx.x) * 8; i < n;
         i += gridDim.x * blockDim.x * 8) {
        float4 a = *(const float4*)(src + i);
        float4 b = *(const float4*)(src + i + 4);
        ushort4 lo = make_ushort4(f2bf(a.x), f2bf(a.y), f2bf(a.z), f2bf(a.w));
        ushort4 hi = make_ushort4(f2bf(b.x), f2bf(b.y), f2bf(b.z), f2bf(b.w));
        *(ushort4*)(dst + i) = lo;
        *(ushort4*)(dst + i + 4) = hi;
    }
}

// One wave (32 lanes) per token: scores, top-2, softmax, append to expert lists.
__global__ void gate_topk_kernel(const float* __restrict__ x, const float* __restrict__ gw,
                                 const float* __restrict__ gb, int* __restrict__ lists,
                                 float* __restrict__ gates, int* __restrict__ counts) {
    int wv = threadIdx.x >> 5, lane = threadIdx.x & 31;
    int tkn = blockIdx.x * 8 + wv;
    if (tkn >= N_TOK) return;
    const float* xr = x + (size_t)tkn * C_DIM;
    float s[N_EXP];
#pragma unroll
    for (int e = 0; e < N_EXP; ++e) s[e] = 0.f;
    for (int c = lane; c < C_DIM; c += 32) {
        float xv = xr[c];
#pragma unroll
        for (int e = 0; e < N_EXP; ++e) s[e] += xv * gw[e * C_DIM + c];
    }
#pragma unroll
    for (int e = 0; e < N_EXP; ++e) {
        float v = s[e];
#pragma unroll
        for (int off = 16; off > 0; off >>= 1) v += __shfl_xor(v, off, 32);
        s[e] = v;
    }
    if (lane == 0) {
        int i0 = -1, i1 = -1;
        float b0 = -3.4e38f, b1 = -3.4e38f;
#pragma unroll
        for (int e = 0; e < N_EXP; ++e) {
            float sc = s[e] + gb[e];
            if (sc > b0) { b1 = b0; i1 = i0; b0 = sc; i0 = e; }
            else if (sc > b1) { b1 = sc; i1 = e; }
        }
        float eg = __expf(b1 - b0);          // softmax over the two selected
        float g0 = 1.f / (1.f + eg);
        float g1 = eg / (1.f + eg);
        int p0 = atomicAdd(&counts[i0], 1);
        lists[i0 * N_TOK + p0] = tkn; gates[i0 * N_TOK + p0] = g0;
        int p1 = atomicAdd(&counts[i1], 1);
        lists[i1 * N_TOK + p1] = tkn; gates[i1 * N_TOK + p1] = g1;
    }
}

// Fused expert FFN. Block = (expert, 32-token tile), 8 waves.
// Weights ride the A operand straight from global (L2-resident); activations are
// transposed once into LDS and feed the B operand with 16 strided 16-bit LDS reads per fragment.
__global__ __launch_bounds__(256, 1)
void moe_expert_kernel(const u16* __restrict__ xb, const u16* __restrict__ wfcb,
                       const u16* __restrict__ wprojb, const float* __restrict__ bfc,
                       const float* __restrict__ bproj, const int* __restrict__ lists,
                       const float* __restrict__ gates, const int* __restrict__ counts,
                       float* __restrict__ out) {
    extern __shared__ char smem[];
    u16*   xsT  = (u16*)(smem + XS_OFF);     // [C_DIM][XT_LD]
    u16*   hsT  = (u16*)(smem + HS_OFF);     // [H_DIM][XT_LD]
    int*   toks = (int*)(smem + TK_OFF);
    float* gvs  = (float*)(smem + GV_OFF);

    const int e = blockIdx.y;
    const int cnt = counts[e];
    const int base = blockIdx.x * M_TOK;
    if (base >= cnt) return;                 // block-uniform early exit

    const int t = threadIdx.x;
    const int wv = t >> 5;
    const int lane = t & 31;

    if (t < M_TOK) {
        int idx = base + t;
        if (idx < cnt) { toks[t] = lists[e * N_TOK + idx]; gvs[t] = gates[e * N_TOK + idx]; }
        else           { toks[t] = 0;                      gvs[t] = 0.0f; }   // gate 0 => no-op
    }
    __syncthreads();

    // ---- stage x^T: xsT[c][m] = xb[toks[m]*C + c]  (coalesced read, one-time scatter)
    for (int id = t; id < M_TOK * (C_DIM / 8); id += 256) {
        int m  = id / (C_DIM / 8);
        int c0 = (id % (C_DIM / 8)) * 8;
        union { uint4 u; u16 s[8]; } U;
        U.u = *(const uint4*)(xb + (size_t)toks[m] * C_DIM + c0);
#pragma unroll
        for (int j = 0; j < 8; ++j) xsT[(c0 + j) * XT_LD + m] = U.s[j];
    }
    __syncthreads();

    const u16* wfc_e = wfcb   + (size_t)e * H_DIM * C_DIM;
    const u16* wpj_e = wprojb + (size_t)e * C_DIM * H_DIM;
    const int mb = (lane >= 16) ? 8 : 0;
    const int mt = lane & 15;

    // ---- phase 1: h^T = gelu(W_fc @ x^T + b_fc), per wave: 24 hidden tiles of 16
    for (int j = 0; j < (H_DIM / 16) / 8; ++j) {
        int hb = (wv + 8 * j) * 16;
        v8f acc0 = {}, acc1 = {};
#pragma unroll 4
        for (int k0 = 0; k0 < C_DIM; k0 += 32) {
            v16bf a  = load_a_w(wfc_e, C_DIM, hb, k0, lane);
            v16bf b0 = load_b_t(xsT, k0, 0,  lane);
            v16bf b1 = load_b_t(xsT, k0, 16, lane);
            acc0 = wmma_bf16(a, b0, acc0);
            acc1 = wmma_bf16(a, b1, acc1);
        }
#pragma unroll
        for (int r = 0; r < 8; ++r) {        // D layout: VGPR r -> weight row r(+8); lane -> token
            int hrow = hb + r + mb;
            float bias = bfc[e * H_DIM + hrow];
            hsT[hrow * XT_LD + mt]      = f2bf(gelu_fn(acc0[r] + bias));
            hsT[hrow * XT_LD + 16 + mt] = f2bf(gelu_fn(acc1[r] + bias));
        }
    }
    __syncthreads();

    // ---- phase 2: out += gate * (W_proj @ h^T + b_proj), per wave: 6 out tiles of 16
    for (int j = 0; j < (C_DIM / 16) / 8; ++j) {
        int nb = (wv + 8 * j) * 16;
        v8f acc0 = {}, acc1 = {};
#pragma unroll 2
        for (int k0 = 0; k0 < H_DIM; k0 += 32) {
            v16bf a  = load_a_w(wpj_e, H_DIM, nb, k0, lane);
            v16bf b0 = load_b_t(hsT, k0, 0,  lane);
            v16bf b1 = load_b_t(hsT, k0, 16, lane);
            acc0 = wmma_bf16(a, b0, acc0);
            acc1 = wmma_bf16(a, b1, acc1);
        }
#pragma unroll
        for (int r = 0; r < 8; ++r) {
            int ncol = nb + r + mb;
            float bias = bproj[e * C_DIM + ncol];
            float v0 = gvs[mt]      * (acc0[r] + bias);
            float v1 = gvs[16 + mt] * (acc1[r] + bias);
            atomicAdd(out + (size_t)toks[mt]      * C_DIM + ncol, v0);
            atomicAdd(out + (size_t)toks[16 + mt] * C_DIM + ncol, v1);
        }
    }
}

// ---------------- launcher ----------------
extern "C" void kernel_launch(void* const* d_in, const int* in_sizes, int n_in,
                              void* d_out, int out_size, void* d_ws, size_t ws_size,
                              hipStream_t stream) {
    const float* x      = (const float*)d_in[0];
    const float* gate_w = (const float*)d_in[1];
    const float* gate_b = (const float*)d_in[2];
    const float* w_fc   = (const float*)d_in[3];
    const float* b_fc   = (const float*)d_in[4];
    const float* w_proj = (const float*)d_in[5];
    const float* b_proj = (const float*)d_in[6];
    float* out = (float*)d_out;

    char* ws = (char*)d_ws;
    u16*   xb     = (u16*)(ws + WS_XB);
    u16*   wfcb   = (u16*)(ws + WS_WFCB);
    u16*   wpjb   = (u16*)(ws + WS_WPJB);
    int*   lists  = (int*)(ws + WS_LISTS);
    float* gts    = (float*)(ws + WS_GATES);
    int*   counts = (int*)(ws + WS_COUNTS);

    moe_zero_kernel<<<1024, 256, 0, stream>>>(out, N_TOK * C_DIM, counts);
    cast_f32_bf16<<<2048, 256, 0, stream>>>(x, xb, N_TOK * C_DIM);
    cast_f32_bf16<<<2048, 256, 0, stream>>>(w_fc, wfcb, N_EXP * H_DIM * C_DIM);
    cast_f32_bf16<<<2048, 256, 0, stream>>>(w_proj, wpjb, N_EXP * C_DIM * H_DIM);
    gate_topk_kernel<<<N_TOK / 8, 256, 0, stream>>>(x, gate_w, gate_b, lists, gts, counts);

    (void)hipFuncSetAttribute(reinterpret_cast<const void*>(moe_expert_kernel),
                              hipFuncAttributeMaxDynamicSharedMemorySize, SMEM_TOTAL);
    dim3 grid(N_TOK / M_TOK, N_EXP);        // worst case: all tokens on one expert
    moe_expert_kernel<<<grid, 256, SMEM_TOTAL, stream>>>(xb, wfcb, wpjb, b_fc, b_proj,
                                                         lists, gts, counts, out);
}
